// PatchEncoder_21345987461199
// MI455X (gfx1250) — compile-verified, matrix operand order checked by reading, not run on hardware
//
#include <hip/hip_runtime.h>
#include <hip/hip_bf16.h>
#include <cstdint>

// ---------------------------------------------------------------------------
// Bidirectional Mamba patch encoder for MI455X (gfx1250, wave32).
// - GEMMs: v_wmma_f32_16x16x32_bf16, 16x64 strip/wave, both operands bf16
//   (weights converted once; activations mirrored by their producers), so
//   the inner loop is pure b128 loads + wmma.  Epilogue variants are
//   template-pruned (ACT / has-f32-out / has-bf16-out).
// - Selective scan: 3-pass segmented linear scan (32 segments) — serial
//   depth ~160 instead of 2049, 32K threads instead of 1K.  Exact because
//   the segment propagator is exp(A * sum(dt)).
// ---------------------------------------------------------------------------

namespace {

constexpr int kB   = 8;
constexpr int kN   = 16;
constexpr int kP   = 128;
constexpr int kPL  = 16;
constexpr int kD   = 128;
constexpr int kDS  = 16;
constexpr int kDTR = 8;
constexpr int kDFF = 256;
constexpr int kEL  = 2;
constexpr int kL   = kN * kP + 1;   // 2049 tokens (incl. view token)
constexpr int kLp  = 2064;          // padded to multiple of 16 (129 tiles)
constexpr int kRows = kB * kLp;     // 16512 GEMM rows (multiple of 16)
constexpr int kSeg = 32;            // scan segments
constexpr int kLs  = (kL + kSeg - 1) / kSeg;   // 65 steps per segment

typedef __attribute__((ext_vector_type(16))) __bf16 v16bf;
typedef __attribute__((ext_vector_type(8)))  float  v8f;

__device__ __forceinline__ unsigned short f2bf(float f) {
  union { float f; unsigned u; } c; c.f = f;
  unsigned x = c.u;
  unsigned r = x + 0x7FFFu + ((x >> 16) & 1u);  // RNE
  return (unsigned short)(r >> 16);
}

__device__ __forceinline__ float siluf(float x) {
  return x / (1.f + __expf(-x));
}

union Frag { v16bf v; uint4 q[2]; };

// ---------------------------------------------------------------------------
// Strip GEMM: out[M, 16*NT*nstrips] = act(A[M,K] @ Wbf[N,K]^T + bias)
// A, W bf16 row-major K-contiguous; M%16==0; K in {16,32k}; 16B alignment
// by construction.  ACT: 0=none, 1=relu, 2=softplus.  HASC: f32 output
// (accum => +=).  HASCB: bf16 mirror output.
// ---------------------------------------------------------------------------
template <int K, int NT, int ACT, bool HASC, bool HASCB>
__global__ void gemm_wmma_bf16(const unsigned short* __restrict__ A, int lda,
                               const unsigned short* __restrict__ Wb, int ldw,
                               const float* __restrict__ bias,
                               float* __restrict__ C,
                               unsigned short* __restrict__ Cb, int ldc,
                               int M, int nstrips, int accum) {
  int wave = (int)((blockIdx.x * blockDim.x + threadIdx.x) >> 5);
  int lane = threadIdx.x & 31;
  int mt = wave / nstrips;
  int st = wave - mt * nstrips;
  if (mt * 16 >= M) return;

  int half16 = lane >> 4;   // selects K sub-range per ISA fragment layout
  int l15    = lane & 15;

  const unsigned short* Ap = A + (size_t)(mt * 16 + l15) * lda;
  const unsigned short* Wp[NT];
#pragma unroll
  for (int j = 0; j < NT; ++j)
    Wp[j] = Wb + (size_t)((st * NT + j) * 16 + l15) * ldw;

  v8f acc[NT];
#pragma unroll
  for (int j = 0; j < NT; ++j) acc[j] = v8f{0.f,0.f,0.f,0.f,0.f,0.f,0.f,0.f};

  constexpr int KC = (K + 31) / 32;
#pragma unroll
  for (int c = 0; c < KC; ++c) {
    const int kc = c * 32;
    const bool v0 = (kc + 16) <= K;   // chunk0: frag elems 0..7
    const bool v1 = (kc + 32) <= K;   // chunk1: frag elems 8..15
    int kb = kc + half16 * 8;

    Frag af;
    if (v0) af.q[0] = *reinterpret_cast<const uint4*>(Ap + kb);
    else    af.q[0] = uint4{0u, 0u, 0u, 0u};
    if (v1) af.q[1] = *reinterpret_cast<const uint4*>(Ap + kb + 16);
    else    af.q[1] = uint4{0u, 0u, 0u, 0u};

#pragma unroll
    for (int j = 0; j < NT; ++j) {
      Frag bf;
      if (v0) bf.q[0] = *reinterpret_cast<const uint4*>(Wp[j] + kb);
      else    bf.q[0] = uint4{0u, 0u, 0u, 0u};
      if (v1) bf.q[1] = *reinterpret_cast<const uint4*>(Wp[j] + kb + 16);
      else    bf.q[1] = uint4{0u, 0u, 0u, 0u};
      acc[j] = __builtin_amdgcn_wmma_f32_16x16x32_bf16(
          false, af.v, false, bf.v, (short)0, acc[j], false, false);
    }
  }

#pragma unroll
  for (int j = 0; j < NT; ++j) {
    int col = (st * NT + j) * 16 + l15;
    float bv = bias ? bias[col] : 0.f;
#pragma unroll
    for (int v = 0; v < 8; ++v) {
      int row = mt * 16 + half16 * 8 + v;   // ISA f32 C/D fragment row map
      float r = acc[j][v] + bv;
      if (ACT == 1) r = fmaxf(r, 0.f);
      if (ACT == 2) r = (r > 20.f) ? r : log1pf(__expf(r));
      size_t ci = (size_t)row * ldc + col;
      if (HASC)  { if (accum) C[ci] += r; else C[ci] = r; }
      if (HASCB) Cb[ci] = f2bf(r);
    }
  }
}

// ---------------------------------------------------------------------------
// f32 -> bf16 convert with optional zero padding (weights + x input).
// ---------------------------------------------------------------------------
__global__ void cvt_pad_bf16(unsigned short* __restrict__ dst,
                             const float* __restrict__ src,
                             int rowsD, int RbD, int RbS, int Kd, int Ks) {
  int idx = blockIdx.x * blockDim.x + threadIdx.x;
  if (idx >= rowsD * Kd) return;
  int r = idx / Kd;
  int k = idx - r * Kd;
  int blk = r / RbD, rr = r - blk * RbD;
  float v = (rr < RbS && k < Ks) ? src[((size_t)blk * RbS + rr) * Ks + k] : 0.f;
  dst[idx] = f2bf(v);
}

// ---------------------------------------------------------------------------
// u assembly: patch tokens + view token + zero padding rows. f32 + bf16.
// ---------------------------------------------------------------------------
__global__ void assemble_u(const float* __restrict__ tmp,
                           const float* __restrict__ view,
                           float* __restrict__ u,
                           unsigned short* __restrict__ ub) {
  int idx = blockIdx.x * blockDim.x + threadIdx.x;
  if (idx >= kRows * kD) return;
  int d = idx & (kD - 1);
  int tok = idx >> 7;
  int b = tok / kLp;
  int t = tok - b * kLp;
  float v;
  if (t < kN * kP)        v = tmp[((size_t)b * (kN * kP) + t) * kD + d];
  else if (t == kN * kP)  v = view[b * kD + d];
  else                    v = 0.f;
  u[idx]  = v;
  ub[idx] = f2bf(v);
}

// ---------------------------------------------------------------------------
// Depthwise conv (width 2, direction-aware neighbor) + SiLU. f32 + bf16.
// ---------------------------------------------------------------------------
__global__ void conv2_silu(const float* __restrict__ xz,
                           const float* __restrict__ cw,   // (128,2)
                           const float* __restrict__ cb,   // (128)
                           float* __restrict__ xcs,
                           unsigned short* __restrict__ xcsb, int dir) {
  int idx = blockIdx.x * blockDim.x + threadIdx.x;
  if (idx >= kRows * kD) return;
  int d = idx & (kD - 1);
  int tok = idx >> 7;
  int b = tok / kLp;
  int t = tok - b * kLp;
  float r = 0.f;
  if (t < kL) {
    int tn = dir ? (t + 1) : (t - 1);
    float prev = (tn >= 0 && tn < kL) ? xz[((size_t)b * kLp + tn) * (2 * kD) + d] : 0.f;
    float cur  = xz[((size_t)b * kLp + t) * (2 * kD) + d];
    float v = prev * cw[d * 2 + 0] + cur * cw[d * 2 + 1] + cb[d];
    r = siluf(v);
  }
  xcs[idx]  = r;
  xcsb[idx] = f2bf(r);
}

// ---------------------------------------------------------------------------
// Segmented selective scan.  Per (b, dir, d, n):
//   h_t = exp(dt_t * A) * h_{t-1} + dt_t * Bm_t * xc_t ;  y_t = <h_t, Cm_t>
// Pass 1: per-segment terminal state from h=0, plus sum(dt) (propagator
//         is exp(A * sum(dt))).  Grid: B*Seg blocks x 128 threads.
// Pass 2: 32-step sequential fix-up producing each segment's initial state.
// Pass 3: re-run segments from true initial state, emit gated bf16 output.
// ---------------------------------------------------------------------------
__global__ void scan_pass1(const float* __restrict__ dt,
                           const float* __restrict__ dbc,
                           const float* __restrict__ xcs,
                           const float* __restrict__ Alog,
                           float* __restrict__ Tst,   // (B*Seg,128,16)
                           float* __restrict__ sdt,   // (B*Seg,128)
                           int dir) {
  int b = blockIdx.x / kSeg;
  int sg = blockIdx.x - b * kSeg;
  int d = threadIdx.x;
  float A[kDS];
#pragma unroll
  for (int n = 0; n < kDS; ++n) A[n] = -__expf(Alog[d * kDS + n]);
  float h[kDS];
#pragma unroll
  for (int n = 0; n < kDS; ++n) h[n] = 0.f;
  float sum = 0.f;
  int s0 = sg * kLs, s1 = min(s0 + kLs, kL);
  for (int si = s0; si < s1; ++si) {
    int t = dir ? (kL - 1 - si) : si;
    size_t rb = (size_t)b * kLp + t;
    float dtv = dt[rb * kD + d];
    float xv  = xcs[rb * kD + d];
    const float* dbcr = dbc + rb * 48;
    float dx = dtv * xv;
    sum += dtv;
#pragma unroll
    for (int n = 0; n < kDS; ++n)
      h[n] = __expf(dtv * A[n]) * h[n] + dx * dbcr[8 + n];
  }
  size_t base = (size_t)(b * kSeg + sg) * kD + d;
  sdt[base] = sum;
#pragma unroll
  for (int n = 0; n < kDS; ++n) Tst[base * kDS + n] = h[n];
}

__global__ void scan_pass2(const float* __restrict__ Tst,
                           const float* __restrict__ sdt,
                           const float* __restrict__ Alog,
                           float* __restrict__ Hinit) {  // (B*Seg,128,16)
  int idx = blockIdx.x * blockDim.x + threadIdx.x;   // B*128*16 = 16384
  if (idx >= kB * kD * kDS) return;
  int b = idx / (kD * kDS);
  int dn = idx - b * (kD * kDS);
  int d = dn >> 4, n = dn & 15;
  float A = -__expf(Alog[d * kDS + n]);
  float H = 0.f;
  for (int sg = 0; sg < kSeg; ++sg) {
    size_t base = (size_t)(b * kSeg + sg) * kD + d;
    Hinit[base * kDS + n] = H;
    H = __expf(A * sdt[base]) * H + Tst[base * kDS + n];
  }
}

__global__ void scan_pass3(const float* __restrict__ dt,
                           const float* __restrict__ dbc,
                           const float* __restrict__ xcs,
                           const float* __restrict__ xz,
                           const float* __restrict__ Alog,
                           const float* __restrict__ Dp,
                           const float* __restrict__ Hinit,
                           unsigned short* __restrict__ ygb,  // (B,Lp,128) bf16
                           int dir) {
  int b = blockIdx.x / kSeg;
  int sg = blockIdx.x - b * kSeg;
  int d = threadIdx.x;
  float A[kDS];
#pragma unroll
  for (int n = 0; n < kDS; ++n) A[n] = -__expf(Alog[d * kDS + n]);
  float Dd = Dp[d];
  size_t base = (size_t)(b * kSeg + sg) * kD + d;
  float h[kDS];
#pragma unroll
  for (int n = 0; n < kDS; ++n) h[n] = Hinit[base * kDS + n];
  int s0 = sg * kLs, s1 = min(s0 + kLs, kL);
  for (int si = s0; si < s1; ++si) {
    int t = dir ? (kL - 1 - si) : si;
    size_t rb = (size_t)b * kLp + t;
    float dtv = dt[rb * kD + d];
    float xv  = xcs[rb * kD + d];
    const float* dbcr = dbc + rb * 48;
    float dx = dtv * xv;
    float y = 0.f;
#pragma unroll
    for (int n = 0; n < kDS; ++n) {
      h[n] = __expf(dtv * A[n]) * h[n] + dx * dbcr[8 + n];
      y = fmaf(h[n], dbcr[24 + n], y);
    }
    y = fmaf(xv, Dd, y);
    float zv = xz[rb * (2 * kD) + kD + d];
    ygb[rb * kD + d] = f2bf(y * siluf(zv));
  }
}

// zero bf16 yg padding rows once (pass3 never touches t >= kL)
__global__ void zero_yg_pads(unsigned short* __restrict__ ygb) {
  int idx = blockIdx.x * blockDim.x + threadIdx.x;
  int tot = kB * (kLp - kL) * kD;
  if (idx >= tot) return;
  int d = idx & (kD - 1);
  int r = idx >> 7;
  int b = r / (kLp - kL);
  int t = kL + (r - b * (kLp - kL));
  ygb[((size_t)b * kLp + t) * kD + d] = 0;
}

// ---------------------------------------------------------------------------
// LayerNorm over D=128: one wave32 per row, shfl_xor reductions.
// ---------------------------------------------------------------------------
__global__ void layernorm128(const float* __restrict__ a,
                             const float* __restrict__ bopt,
                             const float* __restrict__ g,
                             const float* __restrict__ be,
                             float* __restrict__ out,
                             unsigned short* __restrict__ outb, int M) {
  int wave = (int)((blockIdx.x * blockDim.x + threadIdx.x) >> 5);
  int lane = threadIdx.x & 31;
  if (wave >= M) return;
  const float* ar = a + (size_t)wave * kD;
  const float* br = bopt ? bopt + (size_t)wave * kD : nullptr;
  float v[4];
  float s = 0.f;
#pragma unroll
  for (int i = 0; i < 4; ++i) {
    int c = lane + i * 32;
    v[i] = ar[c] + (br ? br[c] : 0.f);
    s += v[i];
  }
#pragma unroll
  for (int o = 16; o > 0; o >>= 1) s += __shfl_xor(s, o, 32);
  float mean = s * (1.f / kD);
  float vs = 0.f;
#pragma unroll
  for (int i = 0; i < 4; ++i) { float dd = v[i] - mean; vs += dd * dd; }
#pragma unroll
  for (int o = 16; o > 0; o >>= 1) vs += __shfl_xor(vs, o, 32);
  float rstd = rsqrtf(vs * (1.f / kD) + 1e-5f);
#pragma unroll
  for (int i = 0; i < 4; ++i) {
    int c = lane + i * 32;
    float r = (v[i] - mean) * rstd * g[c] + be[c];
    out[(size_t)wave * kD + c] = r;
    if (outb) outb[(size_t)wave * kD + c] = f2bf(r);
  }
}

__global__ void add_inplace(float* __restrict__ u, const float* __restrict__ acc) {
  int idx = blockIdx.x * blockDim.x + threadIdx.x;
  if (idx < kRows * kD) u[idx] += acc[idx];
}

// out[b,n,d,p] = uf[b, n*P + p, d]
__global__ void write_out(const float* __restrict__ uf, float* __restrict__ out) {
  int idx = blockIdx.x * blockDim.x + threadIdx.x;
  if (idx >= kB * kN * kD * kP) return;
  int p = idx & (kP - 1);
  int d = (idx >> 7) & (kD - 1);
  int n = (idx >> 14) & (kN - 1);
  int b = idx >> 18;
  out[idx] = uf[(((size_t)b * kLp) + n * kP + p) * kD + d];
}

template <int K, int NT, int ACT, bool HASC, bool HASCB>
static void launch_gemm(const unsigned short* A, int lda,
                        const unsigned short* W, int ldw, const float* bias,
                        float* C, unsigned short* Cb, int ldc, int M, int N,
                        int accum, hipStream_t s) {
  int nstrips = N / (16 * NT);
  long waves = (long)(M / 16) * nstrips;
  int blocks = (int)((waves + 7) / 8);   // 8 wave32 per 256-thread block
  gemm_wmma_bf16<K, NT, ACT, HASC, HASCB><<<blocks, 256, 0, s>>>(
      A, lda, W, ldw, bias, C, Cb, ldc, M, nstrips, accum);
}

}  // namespace

// ---------------------------------------------------------------------------
// Host orchestration (all launches on `stream`; scratch carved from d_ws).
// ---------------------------------------------------------------------------
extern "C" void kernel_launch(void* const* d_in, const int* in_sizes, int n_in,
                              void* d_out, int out_size, void* d_ws, size_t ws_size,
                              hipStream_t stream) {
  const float* x       = (const float*)d_in[0];
  const float* view    = (const float*)d_in[1];
  const float* wp_w    = (const float*)d_in[2];
  const float* wp_b    = (const float*)d_in[3];
  const float* in_w    = (const float*)d_in[4];
  const float* conv_w  = (const float*)d_in[5];
  const float* conv_b  = (const float*)d_in[6];
  const float* xproj_w = (const float*)d_in[7];
  const float* dt_w    = (const float*)d_in[8];
  const float* dt_b    = (const float*)d_in[9];
  const float* Alog    = (const float*)d_in[10];
  const float* Dp      = (const float*)d_in[11];
  const float* out_w   = (const float*)d_in[12];
  const float* fw1     = (const float*)d_in[13];
  const float* fb1     = (const float*)d_in[14];
  const float* fw2     = (const float*)d_in[15];
  const float* fb2     = (const float*)d_in[16];
  const float* ln1g    = (const float*)d_in[17];
  const float* ln1b    = (const float*)d_in[18];
  const float* ln2g    = (const float*)d_in[19];
  const float* ln2b    = (const float*)d_in[20];
  const float* lnfg    = (const float*)d_in[21];
  const float* lnfb    = (const float*)d_in[22];

  float* ws = (float*)d_ws;
  size_t off = 0;
  float* u    = ws + off; off += (size_t)kRows * kD;       // (B,Lp,128)
  float* acc  = ws + off; off += (size_t)kRows * kD;       // fwd+rev accum / patch tmp
  float* xz   = ws + off; off += (size_t)kRows * 2 * kD;   // (B,Lp,256)
  float* xcs  = ws + off; off += (size_t)kRows * kD;       // conv+silu output
  float* dbc  = ws + off; off += (size_t)kRows * 48;       // x-proj (48-wide, 40 live)
  float* dtb  = ws + off; off += (size_t)kRows * kD;       // softplus dt
  float* yg   = ws + off; off += (size_t)kRows * kD;       // FFN output
  float* xr   = ws + off; off += (size_t)kRows * kD;       // LN1 / final LN
  float* Tst  = ws + off; off += (size_t)kB * kSeg * kD * kDS;  // scan terminals
  float* Hini = ws + off; off += (size_t)kB * kSeg * kD * kDS;  // scan initial states
  float* sdt  = ws + off; off += (size_t)kB * kSeg * kD;        // per-segment sum(dt)

  // bf16 activation mirrors (A-operands for WMMA)
  unsigned short* bw = (unsigned short*)(ws + off);
  size_t boff = 0;
  unsigned short* x_bf    = bw + boff; boff += (size_t)kB*kN*kP * kPL;
  unsigned short* u_bf    = bw + boff; boff += (size_t)kRows * kD;
  unsigned short* xcs_bf  = bw + boff; boff += (size_t)kRows * kD;
  unsigned short* dbc_bf  = bw + boff; boff += (size_t)kRows * 48;
  unsigned short* yg_bf   = bw + boff; boff += (size_t)kRows * kD;
  unsigned short* xr_bf   = bw + boff; boff += (size_t)kRows * kD;
  unsigned short* hid_bf  = (unsigned short*)acc;   // aliases acc (disjoint lifetime)

  // bf16 weight mirrors (padded where ragged)
  unsigned short* wpw_bf   = bw + boff; boff += (size_t)kD * kPL;          // (128,16)
  unsigned short* inw_bf   = bw + boff; boff += (size_t)4 * 2 * kD * kD;   // (4,256,128)
  unsigned short* xprojw_bf= bw + boff; boff += (size_t)4 * 48 * kD;       // (4,48,128)
  unsigned short* dtw_bf   = bw + boff; boff += (size_t)4 * kD * 16;       // (4,128,16)
  unsigned short* outw_bf  = bw + boff; boff += (size_t)4 * kD * kD;       // (4,128,128)
  unsigned short* fw1_bf   = bw + boff; boff += (size_t)kEL * kDFF * kD;   // (2,256,128)
  unsigned short* fw2_bf   = bw + boff; boff += (size_t)kEL * kD * kDFF;   // (2,128,256)

  auto cvt = [&](unsigned short* dst, const float* src, int rowsD, int RbD,
                 int RbS, int Kd, int Ks) {
    int tot = rowsD * Kd;
    cvt_pad_bf16<<<(tot + 255) / 256, 256, 0, stream>>>(dst, src, rowsD, RbD,
                                                        RbS, Kd, Ks);
  };
  cvt(x_bf,      x,       kB*kN*kP,  kB*kN*kP,  kB*kN*kP,  kPL, kPL);
  cvt(wpw_bf,    wp_w,    kD,        kD,        kD,        kPL, kPL);
  cvt(inw_bf,    in_w,    4 * 2*kD,  4 * 2*kD,  4 * 2*kD,  kD,  kD);
  cvt(xprojw_bf, xproj_w, 4 * 48,    48,        40,        kD,  kD);
  cvt(dtw_bf,    dt_w,    4 * kD,    4 * kD,    4 * kD,    16,  kDTR);
  cvt(outw_bf,   out_w,   4 * kD,    4 * kD,    4 * kD,    kD,  kD);
  cvt(fw1_bf,    fw1,     kEL*kDFF,  kEL*kDFF,  kEL*kDFF,  kD,  kD);
  cvt(fw2_bf,    fw2,     kEL*kD,    kEL*kD,    kEL*kD,    kDFF, kDFF);

  const int elemBlocks = (kRows * kD + 255) / 256;
  const int lnBlocks   = (kRows + 7) / 8;
  const int scanBlocks = kB * kSeg;   // 256 blocks x 128 threads

  zero_yg_pads<<<(kB * (kLp - kL) * kD + 255) / 256, 256, 0, stream>>>(yg_bf);

  // Patch projection: (B*N*P,16) @ wp_w^T + wp_b -> acc as tmp (f32)
  launch_gemm<16, 4, 0, true, false>(x_bf, kPL, wpw_bf, kPL, wp_b, acc, nullptr,
                                     kD, kB * kN * kP, kD, 0, stream);
  assemble_u<<<elemBlocks, 256, 0, stream>>>(acc, view, u, u_bf);

  for (int l = 0; l < kEL; ++l) {
    for (int dir = 0; dir < 2; ++dir) {
      int pd = l * 2 + dir;
      // xz = u @ in_w^T   (K=128, N=256); consumed elementwise -> f32 only
      launch_gemm<128, 4, 0, true, false>(u_bf, kD,
          inw_bf + (size_t)pd * 2 * kD * kD, kD, nullptr,
          xz, nullptr, 2 * kD, kRows, 2 * kD, 0, stream);
      conv2_silu<<<elemBlocks, 256, 0, stream>>>(
          xz, conv_w + (size_t)pd * kD * 2, conv_b + (size_t)pd * kD,
          xcs, xcs_bf, dir);
      // dbc = xcs @ xproj_w^T  (N=48 padded); f32 for scan + bf16 for dt-proj
      launch_gemm<128, 3, 0, true, true>(xcs_bf, kD,
          xprojw_bf + (size_t)pd * 48 * kD, kD, nullptr,
          dbc, dbc_bf, 48, kRows, 48, 0, stream);
      // dt = softplus(dbc[:, :8] @ dt_w^T + dt_b)  (K padded to 16)
      launch_gemm<16, 4, 2, true, false>(dbc_bf, 48,
          dtw_bf + (size_t)pd * kD * 16, 16, dt_b + (size_t)pd * kD,
          dtb, nullptr, kD, kRows, kD, 0, stream);
      // segmented scan (3 passes)
      const float* Al = Alog + (size_t)pd * kD * kDS;
      scan_pass1<<<scanBlocks, kD, 0, stream>>>(dtb, dbc, xcs, Al, Tst, sdt, dir);
      scan_pass2<<<(kB * kD * kDS + 255) / 256, 256, 0, stream>>>(Tst, sdt, Al, Hini);
      scan_pass3<<<scanBlocks, kD, 0, stream>>>(dtb, dbc, xcs, xz, Al,
          Dp + (size_t)pd * kD, Hini, yg_bf, dir);
      // out-projection; dir0 writes acc, dir1 accumulates (fwd + rev)
      launch_gemm<128, 4, 0, true, false>(yg_bf, kD,
          outw_bf + (size_t)pd * kD * kD, kD, nullptr,
          acc, nullptr, kD, kRows, kD, dir, stream);
    }
    add_inplace<<<elemBlocks, 256, 0, stream>>>(u, acc);
    layernorm128<<<lnBlocks, 256, 0, stream>>>(u, nullptr,
        ln1g + (size_t)l * kD, ln1b + (size_t)l * kD, xr, xr_bf, kRows);
    // FFN: h = relu(xr @ w1^T + b1) -> bf16 only;  y2 = h @ w2^T + b2 -> f32
    launch_gemm<128, 4, 1, false, true>(xr_bf, kD,
        fw1_bf + (size_t)l * kDFF * kD, kD, fb1 + (size_t)l * kDFF,
        nullptr, hid_bf, kDFF, kRows, kDFF, 0, stream);
    launch_gemm<256, 4, 0, true, false>(hid_bf, kDFF,
        fw2_bf + (size_t)l * kD * kDFF, kDFF, fb2 + (size_t)l * kD,
        yg, nullptr, kD, kRows, kD, 0, stream);
    layernorm128<<<lnBlocks, 256, 0, stream>>>(xr, yg,
        ln2g + (size_t)l * kD, ln2b + (size_t)l * kD, u, u_bf, kRows);
  }

  layernorm128<<<lnBlocks, 256, 0, stream>>>(u, nullptr, lnfg, lnfb,
                                             xr, nullptr, kRows);
  write_out<<<(kB * kN * kD * kP + 255) / 256, 256, 0, stream>>>(xr, (float*)d_out);

  (void)in_sizes; (void)n_in; (void)out_size; (void)ws_size;
}
